// GraphAttentionEmbedding_12300786335880
// MI455X (gfx1250) — matrix operand order repaired and checked by reading, not compiled, not run on hardware
//
#include <hip/hip_runtime.h>
#include <hip/hip_bf16.h>
#include <stdint.h>

// ---------------------------------------------------------------------------
// GraphAttentionEmbedding (2x TransformerConv, heads=2, C=64) for gfx1250.
//   1) f32 -> bf16 convert of node features
//   2) fused WMMA GEMM  [N x 128] @ [128 x 512] -> q|k|v|s (+bias)
//      - B panel (16 KB) staged to LDS via Tensor Data Mover (tensor_load_to_lds)
//      - each wave: 16 rows x 64 cols = 16 x v_wmma_f32_16x16x32_bf16
//   3) edge alpha = dot(q[dst],k[src])/8 per head, ordered-uint atomicMax
//   4) ex = exp(alpha - amax[dst]); atomicAdd denom
//   5) scatter: out[dst] += v[src] * ex/denom   (out preloaded with skip term)
// ---------------------------------------------------------------------------

typedef __attribute__((ext_vector_type(16))) __bf16    v16bf;
typedef __attribute__((ext_vector_type(8)))  float     v8f;
typedef __attribute__((ext_vector_type(4)))  uint32_t  u32x4;
typedef __attribute__((ext_vector_type(4)))  uint32_t  u32v4;
typedef __attribute__((ext_vector_type(8)))  int       i32v8;
typedef __attribute__((ext_vector_type(4)))  int       i32v4;

union Frag16 { u32x4 u[2]; v16bf v; };

__device__ __forceinline__ uint16_t f2bf(float f) {
  uint32_t u = __float_as_uint(f);
  uint32_t r = (u + 0x7FFFu + ((u >> 16) & 1u)) >> 16;   // RNE
  return (uint16_t)r;
}
__device__ __forceinline__ unsigned f2ord(float f) {     // order-preserving uint
  unsigned u = __float_as_uint(f);
  return (u & 0x80000000u) ? ~u : (u | 0x80000000u);
}
__device__ __forceinline__ float ord2f(unsigned u) {
  return (u & 0x80000000u) ? __uint_as_float(u & 0x7FFFFFFFu) : __uint_as_float(~u);
}

// ---------------------------------------------------------------- convert ---
__global__ void k_f32_to_bf16(const float* __restrict__ in,
                              uint16_t* __restrict__ out, int n) {
  int i = blockIdx.x * blockDim.x + threadIdx.x;
  if (i < n) out[i] = f2bf(in[i]);
}

// ------------------------------------------------------------ weight pack ---
// Pack W{q,k,v,s} [128x128 f32 each, row-major (k,n)] into per-lane WMMA-B
// layout as bf16 pairs:  Wpk[((t*4+ks)*32+lane)*8 + j] holds K=(kbase+2j, +1)
// for column n = t*16 + (lane&15), kbase = ks*32 + (lane>>4)*16.
__global__ void k_pack_w(const float* __restrict__ Wq, const float* __restrict__ Wk,
                         const float* __restrict__ Wv, const float* __restrict__ Ws,
                         const float* __restrict__ bq, const float* __restrict__ bk,
                         const float* __restrict__ bv, const float* __restrict__ bs,
                         uint32_t* __restrict__ Wpk, float* __restrict__ bias512) {
  int tid = blockIdx.x * blockDim.x + threadIdx.x;
  if (tid < 32768) {
    int j    = tid & 7;
    int lane = (tid >> 3) & 31;
    int ks   = (tid >> 8) & 3;
    int t    = tid >> 10;
    int ng   = t * 16 + (lane & 15);               // global column 0..511
    int k0   = ks * 32 + (lane >> 4) * 16 + 2 * j; // K index of low half
    const float* W = (ng < 128) ? Wq : (ng < 256) ? Wk : (ng < 384) ? Wv : Ws;
    int col = ng & 127;
    uint32_t lo = f2bf(W[k0 * 128 + col]);
    uint32_t hi = f2bf(W[(k0 + 1) * 128 + col]);
    Wpk[tid] = lo | (hi << 16);
  } else if (tid < 32768 + 512) {
    int g = tid - 32768;
    const float* b = (g < 128) ? bq : (g < 256) ? bk : (g < 384) ? bv : bs;
    bias512[g] = b[g & 127];
  }
}

// --------------------------------------------------------------- WMMA GEMM --
// Block = 256 threads = 8 waves; tile = 128 rows x 64 cols.
// Wave w handles rows [w*16, w*16+16) x all 64 cols (4 col tiles), K=128 in
// four k-steps -> 16 WMMAs per wave. B panel lives in LDS (TDM-staged).
__global__ __launch_bounds__(256) void k_gemm_qkvs(
    const uint16_t* __restrict__ Xb,   // [N x 128] bf16
    const uint32_t* __restrict__ Wpk,  // packed B operands
    const float*    __restrict__ bias512,
    float*          __restrict__ Out,  // [N x 512] f32
    int Nrows) {
  __shared__ __align__(16) uint32_t sB[4096];      // 16 KB B panel

  const uint32_t* gB = Wpk + (size_t)blockIdx.y * 4096;

#if __has_builtin(__builtin_amdgcn_tensor_load_to_lds) && \
    __has_builtin(__builtin_amdgcn_s_wait_tensorcnt)
  if (threadIdx.x < 32) {   // one wave issues the TDM DMA for the whole block
    uint64_t ga = (uint64_t)(uintptr_t)gB;
    // D# group0: count=1 | lds_addr=0 (sB at LDS base) | global_addr | type=2
    u32v4 g0 = { 1u, 0u, (uint32_t)ga,
                 (uint32_t)((ga >> 32) & 0x01FFFFFFu) | 0x80000000u };
    // D# group1: data_size=4B, tensor_dim0=4096, tensor_dim1=1,
    //            tile_dim0=4096, tensor_dim0_stride=4096 (1-D 16 KB copy)
    i32v8 g1 = { 0x00020000, 0x10000000, 0x00010000, 0x10000000,
                 0, 4096, 0, 0 };
    i32v4 g2 = { 0, 0, 0, 0 };
    i32v4 g3 = { 0, 0, 0, 0 };
    i32v8 g4 = { 0, 0, 0, 0, 0, 0, 0, 0 };
    __builtin_amdgcn_tensor_load_to_lds(g0, g1, g2, g3, g4, 0);
    __builtin_amdgcn_s_wait_tensorcnt(0);
  }
#else
  for (int i = threadIdx.x; i < 4096; i += 256) sB[i] = gB[i];
#endif
  // Escape sB and declare "invisible writer" (the TDM engine) so the compiler
  // cannot fold the LDS reads below to undef; keeps AS(3) provenance so the
  // reads still lower to ds_load_b128.
  asm volatile("" :: "v"(&sB[0]) : "memory");
  __syncthreads();

  int lane  = threadIdx.x & 31;
  int w     = threadIdx.x >> 5;
  int Mbase = blockIdx.x * 128 + w * 16;
  int arow  = Mbase + (lane & 15);
  if (arow >= Nrows) arow = Nrows - 1;             // clamp (store is guarded)
  int khalf = lane >> 4;
  const uint16_t* aptr = Xb + (size_t)arow * 128 + khalf * 16;

  v8f acc[4] = {};
#pragma unroll
  for (int ks = 0; ks < 4; ++ks) {
    Frag16 A;
    A.u[0] = *(const u32x4*)(aptr + ks * 32);        // K = ks*32 + khalf*16 ..
    A.u[1] = *(const u32x4*)(aptr + ks * 32 + 8);    // .. and +8
#pragma unroll
    for (int ct = 0; ct < 4; ++ct) {
      Frag16 B;
      const uint32_t* bp = sB + ((ct * 4 + ks) * 32 + lane) * 8;
      B.u[0] = *(const u32x4*)(bp);                  // ds_load_b128
      B.u[1] = *(const u32x4*)(bp + 4);
      acc[ct] = __builtin_amdgcn_wmma_f32_16x16x32_bf16(
          false, A.v, false, B.v, (short)0, acc[ct], false, false);
    }
  }

  int rbase = Mbase + khalf * 8;
  bool full = (Mbase + 16) <= Nrows;               // wave-uniform fast path
#pragma unroll
  for (int ct = 0; ct < 4; ++ct) {
    int col   = (blockIdx.y * 4 + ct) * 16 + (lane & 15);
    float bia = bias512[col];
    float* op = Out + (size_t)rbase * 512 + col;
    if (full) {
#pragma unroll
      for (int r = 0; r < 8; ++r) op[(size_t)r * 512] = acc[ct][r] + bia;
    } else {
#pragma unroll
      for (int r = 0; r < 8; ++r)
        if (rbase + r < Nrows) op[(size_t)r * 512] = acc[ct][r] + bia;
    }
  }
}

// ----------------------------------------------------- init out with skip ---
__global__ void k_init_out(const float* __restrict__ qkvs,
                           float* __restrict__ out, int n /* N*128 */) {
  int i = blockIdx.x * blockDim.x + threadIdx.x;
  if (i < n) {
    int row = i >> 7, c = i & 127;
    out[i] = qkvs[(size_t)row * 512 + 384 + c];      // x@Ws + bs
  }
}

// ------------------------------------------------------------ edge alpha ----
__global__ void k_edge_alpha(const int* __restrict__ src, const int* __restrict__ dst,
                             const float* __restrict__ qkvs,
                             float* __restrict__ alpha, unsigned* __restrict__ amaxOrd,
                             int E2) {
  int t = blockIdx.x * blockDim.x + threadIdx.x;
  if (t >= E2) return;
  int e = t >> 1, h = t & 1;
  int d = dst[e], s = src[e];
  const float4* qp = (const float4*)(qkvs + (size_t)d * 512 + h * 64);
  const float4* kp = (const float4*)(qkvs + (size_t)s * 512 + 128 + h * 64);
  __builtin_prefetch(qkvs + (size_t)s * 512 + 256, 0, 1);   // warm v[src] for scatter
  float acc = 0.f;
#pragma unroll
  for (int i = 0; i < 16; ++i) {
    float4 a = qp[i], b = kp[i];
    acc += a.x * b.x + a.y * b.y + a.z * b.z + a.w * b.w;
  }
  acc *= 0.125f;                                     // 1/sqrt(64)
  alpha[t] = acc;
  atomicMax(&amaxOrd[d * 2 + h], f2ord(acc));
}

// -------------------------------------------------------------- edge exp ----
__global__ void k_edge_exp(const int* __restrict__ dst, float* __restrict__ alpha,
                           const unsigned* __restrict__ amaxOrd,
                           float* __restrict__ denom, int E2) {
  int t = blockIdx.x * blockDim.x + threadIdx.x;
  if (t >= E2) return;
  int e = t >> 1, h = t & 1;
  int d = dst[e];
  float ex = __expf(alpha[t] - ord2f(amaxOrd[d * 2 + h]));
  alpha[t] = ex;
  atomicAdd(&denom[d * 2 + h], ex);
}

// ---------------------------------------------------------- edge scatter ----
// One wave per edge: 32 lanes x float4 = 128 channels.
__global__ __launch_bounds__(256) void k_edge_scatter(
    const int* __restrict__ src, const int* __restrict__ dst,
    const float* __restrict__ qkvs, const float* __restrict__ alpha,
    const float* __restrict__ denom, float* __restrict__ out, int E) {
  int e = blockIdx.x * 8 + (threadIdx.x >> 5);
  if (e >= E) return;
  int lane = threadIdx.x & 31;
  int d = dst[e], s = src[e];
  int c0 = lane * 4;
  int h  = lane >> 4;                                // channels 0..63 / 64..127
  float a = alpha[e * 2 + h] / denom[d * 2 + h];
  float4 v = *(const float4*)(qkvs + (size_t)s * 512 + 256 + c0);
  float* o = out + (size_t)d * 128 + c0;
  atomicAdd(o + 0, v.x * a);
  atomicAdd(o + 1, v.y * a);
  atomicAdd(o + 2, v.z * a);
  atomicAdd(o + 3, v.w * a);
}

// ------------------------------------------------------------------- relu ---
__global__ void k_relu(float* __restrict__ x, int n) {
  int i = blockIdx.x * blockDim.x + threadIdx.x;
  if (i < n) x[i] = fmaxf(x[i], 0.f);
}

// ---------------------------------------------------------------------------
extern "C" void kernel_launch(void* const* d_in, const int* in_sizes, int n_in,
                              void* d_out, int out_size, void* d_ws, size_t ws_size,
                              hipStream_t stream) {
  const float* x  = (const float*)d_in[0];
  const int*   ei = (const int*)d_in[1];
  const int N = in_sizes[0] / 128;
  const int E = in_sizes[1] / 2;
  const int* src = ei;
  const int* dst = ei + E;

  const float* Wl[2][4], *bl[2][4];
  for (int L = 0; L < 2; ++L)
    for (int m = 0; m < 4; ++m) {
      Wl[L][m] = (const float*)d_in[2 + L * 8 + m * 2];
      bl[L][m] = (const float*)d_in[3 + L * 8 + m * 2];
    }

  // -------- workspace carve-out --------
  size_t off = 0;
  char* base = (char*)d_ws;
  auto carve = [&](size_t bytes) -> void* {
    void* p = base + off;
    off = (off + bytes + 255) & ~(size_t)255;
    return p;
  };
  uint16_t* xb    = (uint16_t*)carve((size_t)N * 128 * 2);
  float*    qkvs  = (float*)   carve((size_t)N * 512 * 4);
  float*    hbuf  = (float*)   carve((size_t)N * 128 * 4);
  float*    alpha = (float*)   carve((size_t)E * 2 * 4);
  unsigned* amax  = (unsigned*)carve((size_t)N * 2 * 4);
  float*    denom = (float*)   carve((size_t)N * 2 * 4);
  uint32_t* wpk1  = (uint32_t*)carve(32768 * 4);
  uint32_t* wpk2  = (uint32_t*)carve(32768 * 4);
  float*    bias1 = (float*)   carve(512 * 4);
  float*    bias2 = (float*)   carve(512 * 4);
  (void)ws_size; (void)n_in; (void)out_size;

  const int TPB = 256;
  dim3 gemm_grid((N + 127) / 128, 8);
  int n_feat = N * 128;

  // -------- pack both layers' weights --------
  int pack_threads = 32768 + 512;
  hipLaunchKernelGGL(k_pack_w, dim3((pack_threads + TPB - 1) / TPB), dim3(TPB), 0, stream,
                     Wl[0][0], Wl[0][1], Wl[0][2], Wl[0][3],
                     bl[0][0], bl[0][1], bl[0][2], bl[0][3], wpk1, bias1);
  hipLaunchKernelGGL(k_pack_w, dim3((pack_threads + TPB - 1) / TPB), dim3(TPB), 0, stream,
                     Wl[1][0], Wl[1][1], Wl[1][2], Wl[1][3],
                     bl[1][0], bl[1][1], bl[1][2], bl[1][3], wpk2, bias2);

  for (int L = 0; L < 2; ++L) {
    const float* feat_in = (L == 0) ? x : hbuf;
    float* layer_out     = (L == 0) ? hbuf : (float*)d_out;
    uint32_t* wpk        = (L == 0) ? wpk1 : wpk2;
    float* bias          = (L == 0) ? bias1 : bias2;

    hipLaunchKernelGGL(k_f32_to_bf16, dim3((n_feat + TPB - 1) / TPB), dim3(TPB), 0, stream,
                       feat_in, xb, n_feat);
    (void)hipMemsetAsync(amax, 0, (size_t)N * 2 * 4, stream);
    (void)hipMemsetAsync(denom, 0, (size_t)N * 2 * 4, stream);

    hipLaunchKernelGGL(k_gemm_qkvs, gemm_grid, dim3(TPB), 0, stream,
                       xb, wpk, bias, qkvs, N);

    hipLaunchKernelGGL(k_init_out, dim3((n_feat + TPB - 1) / TPB), dim3(TPB), 0, stream,
                       qkvs, layer_out, n_feat);

    int E2 = E * 2;
    hipLaunchKernelGGL(k_edge_alpha, dim3((E2 + TPB - 1) / TPB), dim3(TPB), 0, stream,
                       src, dst, qkvs, alpha, amax, E2);
    hipLaunchKernelGGL(k_edge_exp, dim3((E2 + TPB - 1) / TPB), dim3(TPB), 0, stream,
                       dst, alpha, amax, denom, E2);
    hipLaunchKernelGGL(k_edge_scatter, dim3((E + 7) / 8), dim3(TPB), 0, stream,
                       src, dst, qkvs, alpha, denom, layer_out, E);

    if (L == 0)
      hipLaunchKernelGGL(k_relu, dim3((n_feat + TPB - 1) / TPB), dim3(TPB), 0, stream,
                         hbuf, n_feat);
  }
}
